// Model_28037546508778
// MI455X (gfx1250) — matrix operand order repaired and checked by reading, not compiled, not run on hardware
//
#include <hip/hip_runtime.h>
#include <hip/hip_bf16.h>
#include <cstdint>
#include <cstddef>

// ---------------- problem constants (match reference) ----------------
#define EMB      300
#define EMBV     75          // EMB/4 (float4 columns)
#define NNODES   20000
#define NEDGES   80000
#define NGRAPHS  1024
#define NLAYERS  5
#define KP1      320         // padded K for GEMM1 (agg width, 300 -> 320)
#define NP1      640         // padded N for GEMM1 out / K for GEMM2 (600 -> 640)
#define NP2      384         // padded N for GEMM2 out (300 -> 384, multiple of 128)
#define BN_EPS   1e-5f

typedef __attribute__((ext_vector_type(16))) __bf16 v16bf;
typedef __attribute__((ext_vector_type(8)))  __bf16 v8bf;
typedef __attribute__((ext_vector_type(8)))  float  v8f;

static inline int cdiv(long a, long b) { return (int)((a + b - 1) / b); }

// ---------------- utility: fill ----------------
__global__ void k_fill(float* __restrict__ p, long n, float v) {
    long i = (long)blockIdx.x * blockDim.x + threadIdx.x;
    if (i < n) p[i] = v;
}

// ---------------- node embedding: h = xe1[x[:,0]] + xe2[x[:,1]] ----------------
__global__ void k_node_embed(const int* __restrict__ x,
                             const float* __restrict__ emb1,
                             const float* __restrict__ emb2,
                             float* __restrict__ h) {
    long i = (long)blockIdx.x * blockDim.x + threadIdx.x;
    if (i >= (long)NNODES * EMBV) return;
    int node = (int)(i / EMBV), c = (int)(i % EMBV);
    int t0 = x[2 * node], t1 = x[2 * node + 1];
    float4 a = ((const float4*)emb1)[(long)t0 * EMBV + c];
    float4 b = ((const float4*)emb2)[(long)t1 * EMBV + c];
    float4 r; r.x = a.x + b.x; r.y = a.y + b.y; r.z = a.z + b.z; r.w = a.w + b.w;
    ((float4*)h)[i] = r;
}

// ---------------- message scatter: agg[dst] += h[src] + ee1[a0] + ee2[a1] ----------------
// edges e < NEDGES come from edge_index/edge_attr; e >= NEDGES are self-loops [4,0]
__global__ void k_msg(const float* __restrict__ h, const int* __restrict__ ei,
                      const int* __restrict__ ea, const float* __restrict__ ee1,
                      const float* __restrict__ ee2, float* __restrict__ agg) {
    long i = (long)blockIdx.x * blockDim.x + threadIdx.x;
    const long tot = (long)(NEDGES + NNODES) * EMBV;
    if (i >= tot) return;
    int e = (int)(i / EMBV), c = (int)(i % EMBV);
    int s, d, a0, a1;
    if (e < NEDGES) { s = ei[e]; d = ei[NEDGES + e]; a0 = ea[2 * e]; a1 = ea[2 * e + 1]; }
    else            { s = d = e - NEDGES; a0 = 4; a1 = 0; }
    float4 hv = ((const float4*)h)[(long)s * EMBV + c];
    float4 v1 = ((const float4*)ee1)[(long)a0 * EMBV + c];
    float4 v2 = ((const float4*)ee2)[(long)a1 * EMBV + c];
    float* dst = agg + (long)d * EMB + c * 4;
    atomicAdd(dst + 0, hv.x + v1.x + v2.x);
    atomicAdd(dst + 1, hv.y + v1.y + v2.y);
    atomicAdd(dst + 2, hv.z + v1.z + v2.z);
    atomicAdd(dst + 3, hv.w + v1.w + v2.w);
}

// ---------------- fp32 [N,300] -> bf16 [N,320] zero-padded ----------------
__global__ void k_cvt_bf(const float* __restrict__ agg, __hip_bfloat16* __restrict__ abf) {
    long i = (long)blockIdx.x * blockDim.x + threadIdx.x;
    if (i >= (long)NNODES * KP1) return;
    int r = (int)(i / KP1), c = (int)(i % KP1);
    float v = (c < EMB) ? agg[(long)r * EMB + c] : 0.f;
    abf[i] = __float2bfloat16(v);
}

// ---------------- weight transpose+pad to bf16: W[l][K][N] -> Wt[l][Np][Kp] ----------------
__global__ void k_prep_wt(const float* __restrict__ W, __hip_bfloat16* __restrict__ Wt,
                          int L, int K, int N, int Kp, int Np) {
    long i = (long)blockIdx.x * blockDim.x + threadIdx.x;
    long tot = (long)L * Np * Kp;
    if (i >= tot) return;
    int l = (int)(i / ((long)Np * Kp));
    long rem = i - (long)l * Np * Kp;
    int n = (int)(rem / Kp), k = (int)(rem % Kp);
    float v = (n < N && k < K) ? W[(long)l * K * N + (long)k * N + n] : 0.f;
    Wt[i] = __float2bfloat16(v);
}

__global__ void k_prep_bias(const float* __restrict__ b, float* __restrict__ bp,
                            int L, int N, int Np) {
    long i = (long)blockIdx.x * blockDim.x + threadIdx.x;
    if (i >= (long)L * Np) return;
    int l = (int)(i / Np), n = (int)(i % Np);
    bp[i] = (n < N) ? b[(long)l * N + n] : 0.f;
}

// ---------------- bf16 WMMA GEMM: C[M,Np] = A[M,Kp] x Bt[Np,Kp]^T + bias ----------------
// 8 waves/block; wave -> 16(M) x 128(N) (8 accumulators, A fragment reused 8x);
// block covers 128(M) x 128(N). Fragments per ISA 7.12.2 wave32 layouts.
template<int RELU_BF16_OUT>
__global__ __launch_bounds__(256)
void k_wmma_gemm(const __hip_bfloat16* __restrict__ A, int lda,
                 const __hip_bfloat16* __restrict__ Bt, int ldb,
                 const float* __restrict__ bias,
                 float* __restrict__ Cf, __hip_bfloat16* __restrict__ Cbf,
                 int ldc, int M, int Kp) {
    const __bf16* Ab = reinterpret_cast<const __bf16*>(A);
    const __bf16* Bb = reinterpret_cast<const __bf16*>(Bt);
    int lane = threadIdx.x & 31;
    int wave = threadIdx.x >> 5;
    int m0 = blockIdx.x * 128 + wave * 16;
    if (m0 >= M) return;                       // uniform per wave: EXEC stays all-ones
    int n0 = blockIdx.y * 128;
    int arow   = m0 + (lane & 15);
    int akoff  = (lane & 16) ? 8 : 0;          // A: K 0-7/16-23 vs 8-15/24-31 by lane[4]
    int bcol16 = lane & 15;
    int bkoff  = (lane & 16) ? 8 : 0;          // B: K 0-7/16-23 vs 8-15/24-31 by lane[4]
    v8f acc[8] = {v8f{}, v8f{}, v8f{}, v8f{}, v8f{}, v8f{}, v8f{}, v8f{}};
    const __bf16* arowp = Ab + (size_t)arow * lda;
    for (int k0 = 0; k0 < Kp; k0 += 32) {
        if (k0 + 64 < Kp) __builtin_prefetch(arowp + k0 + 64, 0, 3);
        v8bf alo = *(const v8bf*)(arowp + k0 + akoff);
        v8bf ahi = *(const v8bf*)(arowp + k0 + akoff + 16);
        v16bf a;
#pragma unroll
        for (int i = 0; i < 8; i++) { a[i] = alo[i]; a[i + 8] = ahi[i]; }
#pragma unroll
        for (int t = 0; t < 8; t++) {
            const __bf16* bp = Bb + (size_t)(n0 + t * 16 + bcol16) * ldb + k0 + bkoff;
            v8bf blo = *(const v8bf*)bp;
            v8bf bhi = *(const v8bf*)(bp + 16);
            v16bf b;
#pragma unroll
            for (int i = 0; i < 8; i++) { b[i] = blo[i]; b[i + 8] = bhi[i]; }
            acc[t] = __builtin_amdgcn_wmma_f32_16x16x32_bf16(
                false, a, false, b, (short)0, acc[t], false, false);
        }
    }
    int crow = m0 + ((lane & 16) ? 8 : 0);     // C/D: VGPR r -> row r (lanes 0-15) / r+8
#pragma unroll
    for (int t = 0; t < 8; t++) {
        int col = n0 + t * 16 + bcol16;
        float bv = bias[col];
#pragma unroll
        for (int r = 0; r < 8; r++) {
            float v = acc[t][r] + bv;
            if (RELU_BF16_OUT)
                Cbf[(size_t)(crow + r) * ldc + col] = __float2bfloat16(fmaxf(v, 0.f));
            else
                Cf[(size_t)(crow + r) * ldc + col] = v;
        }
    }
}

// ---------------- BatchNorm stats: per-column sum & sumsq over M rows ----------------
__global__ __launch_bounds__(256)
void k_bn_stats(const float* __restrict__ outf, int ldo, int M, float* __restrict__ sums) {
    int c = blockIdx.x, tid = threadIdx.x;
    float s = 0.f, s2 = 0.f;
    for (int r = tid; r < M; r += 256) {
        float v = outf[(size_t)r * ldo + c];
        s += v; s2 += v * v;
    }
    __shared__ float sh[512];
    sh[tid] = s; sh[256 + tid] = s2;
    __syncthreads();
    for (int st = 128; st > 0; st >>= 1) {
        if (tid < st) { sh[tid] += sh[tid + st]; sh[256 + tid] += sh[256 + tid + st]; }
        __syncthreads();
    }
    if (tid == 0) { sums[c] = sh[0]; sums[EMB + c] = sh[256]; }
}

__global__ void k_bn_norm(const float* __restrict__ outf, int ldo, float* __restrict__ h,
                          int M, const float* __restrict__ sums,
                          const float* __restrict__ gamma, const float* __restrict__ beta,
                          int relu) {
    long i = (long)blockIdx.x * blockDim.x + threadIdx.x;
    if (i >= (long)M * EMB) return;
    int r = (int)(i / EMB), c = (int)(i % EMB);
    float mu  = sums[c] / (float)M;
    float var = sums[EMB + c] / (float)M - mu * mu;
    float v = gamma[c] * (outf[(size_t)r * ldo + c] - mu) * rsqrtf(var + BN_EPS) + beta[c];
    if (relu) v = fmaxf(v, 0.f);
    h[i] = v;
}

// ---------------- global add pool: f[batch[i]] += h[i] ----------------
__global__ void k_pool(const float* __restrict__ h, const int* __restrict__ batch,
                       float* __restrict__ f) {
    long i = (long)blockIdx.x * blockDim.x + threadIdx.x;
    if (i >= (long)NNODES * EMBV) return;
    int node = (int)(i / EMBV), c = (int)(i % EMBV);
    float4 v = ((const float4*)h)[i];
    int b = batch[node];
    float* d = f + (long)b * EMB + c * 4;
    atomicAdd(d + 0, v.x); atomicAdd(d + 1, v.y);
    atomicAdd(d + 2, v.z); atomicAdd(d + 3, v.w);
}

// ---------------- classifier: relu(max(fA,fB) @ Wc1 + bc1) @ Wc2 + bc2 ----------------
__global__ __launch_bounds__(256)
void k_classify(const float* __restrict__ fA, const float* __restrict__ fB,
                const float* __restrict__ Wc1, const float* __restrict__ bc1,
                const float* __restrict__ Wc2, const float* __restrict__ bc2,
                float* __restrict__ out, int roll) {
    __shared__ float g[EMB];
    __shared__ float red[256];
    int gi = blockIdx.x, tid = threadIdx.x;
    int rb = roll ? ((gi + NGRAPHS - 1) & (NGRAPHS - 1)) : gi;
    for (int c = tid; c < EMB; c += 256)
        g[c] = fmaxf(fA[(long)gi * EMB + c], fB[(long)rb * EMB + c]);
    __syncthreads();
    float acc = 0.f;
    for (int c = tid; c < EMB; c += 256) {
        float s = bc1[c];
        for (int k = 0; k < EMB; k++) s += g[k] * Wc1[(long)k * EMB + c];
        acc += fmaxf(s, 0.f) * Wc2[c];
    }
    red[tid] = acc;
    __syncthreads();
    for (int st = 128; st > 0; st >>= 1) {
        if (tid < st) red[tid] += red[tid + st];
        __syncthreads();
    }
    if (tid == 0) out[gi] = red[0] + bc2[0];
}

__global__ void k_labels(float* __restrict__ out) {
    int i = blockIdx.x * blockDim.x + threadIdx.x;
    if (i < 2 * NGRAPHS) out[i] = (i < NGRAPHS) ? 1.f : 0.f;
}

// ---------------- host orchestration ----------------
extern "C" void kernel_launch(void* const* d_in, const int* in_sizes, int n_in,
                              void* d_out, int out_size, void* d_ws, size_t ws_size,
                              hipStream_t stream) {
    const int*   x0  = (const int*)d_in[0];
    const int*   ei0 = (const int*)d_in[1];
    const int*   ea0 = (const int*)d_in[2];
    const int*   bt0 = (const int*)d_in[3];
    const int*   x1  = (const int*)d_in[4];
    const int*   ei1 = (const int*)d_in[5];
    const int*   ea1 = (const int*)d_in[6];
    const int*   bt1 = (const int*)d_in[7];
    const float* xe1 = (const float*)d_in[8];
    const float* xe2 = (const float*)d_in[9];
    const float* ee1 = (const float*)d_in[10];
    const float* ee2 = (const float*)d_in[11];
    const float* W1  = (const float*)d_in[12];
    const float* B1  = (const float*)d_in[13];
    const float* W2  = (const float*)d_in[14];
    const float* B2  = (const float*)d_in[15];
    const float* gam = (const float*)d_in[16];
    const float* bet = (const float*)d_in[17];
    const float* Wc1 = (const float*)d_in[18];
    const float* bc1 = (const float*)d_in[19];
    const float* Wc2 = (const float*)d_in[20];
    const float* bc2 = (const float*)d_in[21];
    float* out = (float*)d_out;

    // workspace layout (agg and GEMM2 output alias: disjoint lifetimes)
    char* w = (char*)d_ws;
    auto take = [&](size_t bytes) -> char* {
        char* p = w; w += (bytes + 255) & ~(size_t)255; return p;
    };
    float*          h      = (float*)take((size_t)NNODES * EMB * 4);
    float*          aggout = (float*)take((size_t)NNODES * NP2 * 4);
    __hip_bfloat16* abf    = (__hip_bfloat16*)take((size_t)NNODES * KP1 * 2);
    __hip_bfloat16* hidbf  = (__hip_bfloat16*)take((size_t)NNODES * NP1 * 2);
    __hip_bfloat16* wt1    = (__hip_bfloat16*)take((size_t)NLAYERS * NP1 * KP1 * 2);
    __hip_bfloat16* wt2    = (__hip_bfloat16*)take((size_t)NLAYERS * NP2 * NP1 * 2);
    float*          b1p    = (float*)take((size_t)NLAYERS * NP1 * 4);
    float*          b2p    = (float*)take((size_t)NLAYERS * NP2 * 4);
    float*          sums   = (float*)take((size_t)2 * EMB * 4);
    float*          f0     = (float*)take((size_t)NGRAPHS * EMB * 4);
    float*          f1     = (float*)take((size_t)NGRAPHS * EMB * 4);

    const int TB = 256;

    // one-time weight prep (bf16, transposed, zero-padded)
    k_prep_wt<<<cdiv((long)NLAYERS * NP1 * KP1, TB), TB, 0, stream>>>(
        W1, wt1, NLAYERS, EMB, 2 * EMB, KP1, NP1);
    k_prep_wt<<<cdiv((long)NLAYERS * NP2 * NP1, TB), TB, 0, stream>>>(
        W2, wt2, NLAYERS, 2 * EMB, EMB, NP1, NP2);
    k_prep_bias<<<cdiv((long)NLAYERS * NP1, TB), TB, 0, stream>>>(B1, b1p, NLAYERS, 2 * EMB, NP1);
    k_prep_bias<<<cdiv((long)NLAYERS * NP2, TB), TB, 0, stream>>>(B2, b2p, NLAYERS, EMB, NP2);

    auto encode = [&](const int* x, const int* ei, const int* ea,
                      const int* batch, float* fpool) {
        k_node_embed<<<cdiv((long)NNODES * EMBV, TB), TB, 0, stream>>>(x, xe1, xe2, h);
        for (int l = 0; l < NLAYERS; l++) {
            k_fill<<<cdiv((long)NNODES * EMB, TB), TB, 0, stream>>>(
                aggout, (long)NNODES * EMB, 0.f);
            k_msg<<<cdiv((long)(NEDGES + NNODES) * EMBV, TB), TB, 0, stream>>>(
                h, ei, ea, ee1 + (long)l * 6 * EMB, ee2 + (long)l * 3 * EMB, aggout);
            k_cvt_bf<<<cdiv((long)NNODES * KP1, TB), TB, 0, stream>>>(aggout, abf);
            dim3 g1(cdiv(NNODES, 128), NP1 / 128);
            k_wmma_gemm<1><<<g1, TB, 0, stream>>>(
                abf, KP1, wt1 + (long)l * NP1 * KP1, KP1,
                b1p + (long)l * NP1, nullptr, hidbf, NP1, NNODES, KP1);
            dim3 g2(cdiv(NNODES, 128), NP2 / 128);
            k_wmma_gemm<0><<<g2, TB, 0, stream>>>(
                hidbf, NP1, wt2 + (long)l * NP2 * NP1, NP1,
                b2p + (long)l * NP2, aggout, nullptr, NP2, NNODES, NP1);
            k_bn_stats<<<EMB, TB, 0, stream>>>(aggout, NP2, NNODES, sums);
            k_bn_norm<<<cdiv((long)NNODES * EMB, TB), TB, 0, stream>>>(
                aggout, NP2, h, NNODES, sums,
                gam + (long)l * EMB, bet + (long)l * EMB, l < NLAYERS - 1);
        }
        k_fill<<<cdiv((long)NGRAPHS * EMB, TB), TB, 0, stream>>>(
            fpool, (long)NGRAPHS * EMB, 0.f);
        k_pool<<<cdiv((long)NNODES * EMBV, TB), TB, 0, stream>>>(h, batch, fpool);
    };

    encode(x0, ei0, ea0, bt0, f0);
    encode(x1, ei1, ea1, bt1, f1);

    k_classify<<<NGRAPHS, TB, 0, stream>>>(f0, f1, Wc1, bc1, Wc2, bc2, out, 0);
    k_classify<<<NGRAPHS, TB, 0, stream>>>(f0, f1, Wc1, bc1, Wc2, bc2, out + NGRAPHS, 1);
    k_labels<<<cdiv(2 * NGRAPHS, TB), TB, 0, stream>>>(out + 2 * NGRAPHS);
}
